// ShiftedPatchTokenization_11519102287954
// MI455X (gfx1250) — matrix operand order repaired and checked by reading, not compiled, not run on hardware
//
#include <hip/hip_runtime.h>
#include <stdint.h>

// ---------------------------------------------------------------------------
// Shifted Patch Tokenization on MI455X (gfx1250, wave32)
//   Kernel 1 (prep): W (3840x768 f32) -> transposed bf16 hi/lo planes in d_ws
//   Kernel 2 (GEMM): fused shifted-patch gather + bf16x3 WMMA GEMM
//                    + bias + LayerNorm.  M=18432, K=3840, N=768.
//   Block: 64 rows x 768 cols, 512 threads (16 waves), 12 C-tiles per wave,
//   3x v_wmma_f32_16x16x32_bf16 per tile (a_hi*b_hi + a_lo*b_hi + a_hi*b_lo).
//   B staged via GLOBAL_LOAD_ASYNC_TO_LDS_B128 (ASYNCcnt) when available.
// ---------------------------------------------------------------------------

typedef __attribute__((ext_vector_type(16))) __bf16    v16bf;
typedef __attribute__((ext_vector_type(8)))  float     v8f;
typedef __attribute__((ext_vector_type(8)))  uint32_t  v8u;
typedef int v4i __attribute__((vector_size(16)));   // builtin's pointee type

#define KDIM     3840
#define NCOLS    768
#define ROWS     64
#define TILE_K   32
#define HW       384
#define NPH      24
#define PATCHES  576
#define BN       384          // N columns staged per half

#if __has_builtin(__builtin_amdgcn_global_load_async_to_lds_b128)
#define HAS_ASYNC_LDS 1
#else
#define HAS_ASYNC_LDS 0
#endif

__device__ __forceinline__ uint32_t f2bf_bits(float f) {
  uint32_t u = __float_as_uint(f);
  u += 0x7fffu + ((u >> 16) & 1u);        // round-to-nearest-even
  return u >> 16;
}
__device__ __forceinline__ float bf2f(uint32_t h) {
  return __uint_as_float(h << 16);
}
// two consecutive values -> one dword of hi-bf16 pair + one dword of lo pair
__device__ __forceinline__ void pack_pair(float a, float b,
                                          uint32_t& hi, uint32_t& lo) {
  uint32_t ha = f2bf_bits(a), hb = f2bf_bits(b);
  float ra = a - bf2f(ha), rb = b - bf2f(hb);
  hi = ha | (hb << 16);
  lo = f2bf_bits(ra) | (f2bf_bits(rb) << 16);
}

// ======================= prep: W -> transposed bf16 planes ==================
__global__ __launch_bounds__(256) void wprep_kernel(
    const float* __restrict__ Wm,     // (3840, 768) row-major
    uint16_t* __restrict__ Wth,       // (768, 3840) bf16 hi plane
    uint16_t* __restrict__ Wtl) {     // (768, 3840) bf16 lo plane
  int gid = blockIdx.x * 256 + threadIdx.x;   // 0 .. 46079
  int kc  = gid / NCOLS;                      // 0..59  (64-wide K chunk)
  int n   = gid - kc * NCOLS;                 // 0..767 (fastest -> coalesced)
  int k0  = kc * 64;
#pragma unroll
  for (int s = 0; s < 4; ++s) {
    int kb = k0 + s * 16;
    uint32_t hi[8], lo[8];
#pragma unroll
    for (int j = 0; j < 8; ++j) {
      float a = Wm[(size_t)(kb + 2 * j)     * NCOLS + n];
      float b = Wm[(size_t)(kb + 2 * j + 1) * NCOLS + n];
      pack_pair(a, b, hi[j], lo[j]);
    }
    uint4* dh = (uint4*)(Wth + (size_t)n * KDIM + kb);
    uint4* dl = (uint4*)(Wtl + (size_t)n * KDIM + kb);
    dh[0] = make_uint4(hi[0], hi[1], hi[2], hi[3]);
    dh[1] = make_uint4(hi[4], hi[5], hi[6], hi[7]);
    dl[0] = make_uint4(lo[0], lo[1], lo[2], lo[3]);
    dl[1] = make_uint4(lo[4], lo[5], lo[6], lo[7]);
  }
}

// ============================ LDS swizzled access ===========================
// Row = 16 dwords (32 bf16) = 4 chunks of 16B; chunk c lives at ((c+row)&3).
// Loads 2 chunks (16 bf16, ascending K) as one v16bf fragment. No unpacking.
__device__ __forceinline__ v16bf load_frag2(const uint32_t* __restrict__ base,
                                            int row, int ca, int cb) {
  const uint32_t* rp = base + row * 16;
  uint4 u0 = *(const uint4*)(rp + ((ca + row) & 3) * 4);
  uint4 u1 = *(const uint4*)(rp + ((cb + row) & 3) * 4);
  v8u v = {u0.x, u0.y, u0.z, u0.w, u1.x, u1.y, u1.z, u1.w};
  return __builtin_bit_cast(v16bf, v);
}

// copy one plane of a 384-col x 32-k B half into LDS.
// Preferred path: CDNA5 async global->LDS DMA (no VGPR round-trip, ASYNCcnt).
__device__ __forceinline__ void stage_b_plane(uint32_t* __restrict__ Bdst,
                                              const uint16_t* __restrict__ Wt,
                                              int kb, int h, int tid) {
#pragma unroll
  for (int j = 0; j < 3; ++j) {
    int idx = tid + 512 * j;                 // 0..1535
    int nn  = idx >> 2;                      // 0..383
    int q   = idx & 3;                       // chunk
    const uint16_t* src = Wt + (size_t)(h * BN + nn) * KDIM + kb + q * 8;
    uint32_t* dst = Bdst + nn * 16 + ((q + nn) & 3) * 4;
#if HAS_ASYNC_LDS
    __builtin_amdgcn_global_load_async_to_lds_b128((v4i*)src, (v4i*)dst, 0, 0);
#else
    *(uint4*)dst = *(const uint4*)src;
#endif
  }
}

__device__ __forceinline__ void wait_stage_done() {
#if HAS_ASYNC_LDS
#if __has_builtin(__builtin_amdgcn_s_wait_asynccnt)
  __builtin_amdgcn_s_wait_asynccnt(0);
#else
  asm volatile("s_wait_asynccnt 0x0" ::: "memory");
#endif
#endif
}

// ================================ fused GEMM ================================
__global__ __launch_bounds__(512) void spt_fused_kernel(
    const float* __restrict__ X,        // (32,3,384,384)
    const uint16_t* __restrict__ Wth,   // (768,3840) bf16 hi
    const uint16_t* __restrict__ Wtl,   // (768,3840) bf16 lo
    const float* __restrict__ bias,
    const float* __restrict__ gamma,
    const float* __restrict__ beta,
    float* __restrict__ out)            // (32,576,768)
{
  __shared__ uint32_t Bh[BN * 16];      // 24 KB  B hi plane (half of N)
  __shared__ uint32_t Bl[BN * 16];      // 24 KB  B lo plane
  __shared__ uint32_t Ah[ROWS * 16];    // 4 KB   A hi plane
  __shared__ uint32_t Al[ROWS * 16];    // 4 KB   A lo plane
  __shared__ float    red[ROWS][4][2];  // per-row partial sum/sumsq
  __shared__ float    stats[ROWS][2];   // mean, rstd

  const int tid  = threadIdx.x;
  const int lane = tid & 31;
  const int wid  = tid >> 5;     // 0..15
  const int mt   = wid & 3;      // M sub-tile 0..3
  const int wn   = wid >> 2;     // N group 0..3
  const int blk  = blockIdx.x;

  // --- A-gather coordinates: thread -> one row, 4 consecutive k
  const int arow = tid >> 3;              // 0..63
  const int akq  = (tid & 7) * 4;         // 0..28
  const int rg   = blk * ROWS + arow;
  const int bimg = rg / PATCHES;
  const int pp   = rg - bimg * PATCHES;
  const int ph   = pp / NPH;
  const int pw   = pp - ph * NPH;
  const int ac   = akq >> 3;              // A chunk for this thread's 4 k
  const int aoff = (akq >> 1) & 3;        // dword offset inside chunk (0 or 2)
  const int adw  = arow * 16 + ((ac + arow) & 3) * 4 + aoff;

  // --- fragment addressing (CDNA5 wave32 WMMA layouts)
  const int halfsel = lane >> 4;          // 0: lanes 0-15, 1: lanes 16-31
  const int lrow    = lane & 15;
  const int a_row   = mt * 16 + lrow;     // A-matrix row (M)
  const int a_c0    = halfsel * 2;        // A chunks {c0, c0+1}
  const int b_c0    = halfsel * 2;        // B chunks {c0, c0+1}

  v8f c[12];
#pragma unroll
  for (int i = 0; i < 12; ++i) {
    v8f z = {0.f, 0.f, 0.f, 0.f, 0.f, 0.f, 0.f, 0.f};
    c[i] = z;
  }

  for (int kb = 0; kb < KDIM; kb += TILE_K) {
    __syncthreads();   // previous iteration finished reading LDS

    // ---- stage B half 0 first (async DMA proceeds under the A gather)
    stage_b_plane(Bh, Wth, kb, 0, tid);
    stage_b_plane(Bl, Wtl, kb, 0, tid);

    // ---- stage A: gather shifted-patch features, split to hi/lo bf16 planes
    {
      // thread's 4 features share (channel, py); px consecutive
      int d0  = kb + akq;
      int c15 = d0 >> 8;
      int rem = d0 & 255;
      int py  = rem >> 4;
      int px0 = rem & 15;
      int g   = (c15 * 11) >> 5;           // c15 / 3 for 0..14
      int ch  = c15 - g * 3;
      int dy  = (g == 0) ? 0 : ((g <= 2) ? -1 : 1);
      int dx  = (g == 0) ? 0 : (((g & 1) == 1) ? -1 : 1);
      int y   = ph * 16 + py - dy;
      int x0  = pw * 16 + px0 - dx;
      bool yok = (unsigned)y < (unsigned)HW;
      const float* xrow = X + ((size_t)(bimg * 3 + ch) * HW + y) * HW;
      float v[4];
#pragma unroll
      for (int j = 0; j < 4; ++j) {
        int x = x0 + j;
        v[j] = (yok && (unsigned)x < (unsigned)HW) ? xrow[x] : 0.f;
      }
      uint32_t h0, l0, h1, l1;
      pack_pair(v[0], v[1], h0, l0);
      pack_pair(v[2], v[3], h1, l1);
      *(uint2*)(Ah + adw) = make_uint2(h0, h1);
      *(uint2*)(Al + adw) = make_uint2(l0, l1);
    }

    wait_stage_done();
    __syncthreads();

    // ---- A fragments (held across both halves)
    v16bf ahi = load_frag2(Ah, a_row, a_c0, a_c0 + 1);
    v16bf alo = load_frag2(Al, a_row, a_c0, a_c0 + 1);

    // ---- compute half 0: tiles t = wn + 4*ii, ii = 0..5
#pragma unroll
    for (int ii = 0; ii < 6; ++ii) {
      int nn = (wn + 4 * ii) * 16 + lrow;
      v16bf bhi = load_frag2(Bh, nn, b_c0, b_c0 + 1);
      v16bf blo = load_frag2(Bl, nn, b_c0, b_c0 + 1);
      c[ii] = __builtin_amdgcn_wmma_f32_16x16x32_bf16(false, ahi, false, bhi,
                                                      (short)0, c[ii], false, false);
      c[ii] = __builtin_amdgcn_wmma_f32_16x16x32_bf16(false, alo, false, bhi,
                                                      (short)0, c[ii], false, false);
      c[ii] = __builtin_amdgcn_wmma_f32_16x16x32_bf16(false, ahi, false, blo,
                                                      (short)0, c[ii], false, false);
    }
    __syncthreads();

    // ---- stage B half 1 (cols 384..767)
    stage_b_plane(Bh, Wth, kb, 1, tid);
    stage_b_plane(Bl, Wtl, kb, 1, tid);
    wait_stage_done();
    __syncthreads();

#pragma unroll
    for (int ii = 6; ii < 12; ++ii) {
      int nn = (wn + 4 * ii) * 16 + lrow - BN;
      v16bf bhi = load_frag2(Bh, nn, b_c0, b_c0 + 1);
      v16bf blo = load_frag2(Bl, nn, b_c0, b_c0 + 1);
      c[ii] = __builtin_amdgcn_wmma_f32_16x16x32_bf16(false, ahi, false, bhi,
                                                      (short)0, c[ii], false, false);
      c[ii] = __builtin_amdgcn_wmma_f32_16x16x32_bf16(false, alo, false, bhi,
                                                      (short)0, c[ii], false, false);
      c[ii] = __builtin_amdgcn_wmma_f32_16x16x32_bf16(false, ahi, false, blo,
                                                      (short)0, c[ii], false, false);
    }
  }

  // ======================= epilogue: bias + LayerNorm =======================
  // C layout: elem j -> row (mt*16 + j + 8*halfsel), col = t*16 + lrow.
  float s[8], q[8];
#pragma unroll
  for (int j = 0; j < 8; ++j) { s[j] = 0.f; q[j] = 0.f; }

#pragma unroll
  for (int ii = 0; ii < 12; ++ii) {
    int col  = (wn + 4 * ii) * 16 + lrow;
    float bv = bias[col];
#pragma unroll
    for (int j = 0; j < 8; ++j) {
      float v  = c[ii][j] + bv;
      c[ii][j] = v;
      s[j] += v;
      q[j] += v * v;
    }
  }
  // reduce across the 16 lanes of each half-group (masks 1..8 stay in-group)
#pragma unroll
  for (int j = 0; j < 8; ++j) {
#pragma unroll
    for (int m = 1; m <= 8; m <<= 1) {
      s[j] += __shfl_xor(s[j], m, 32);
      q[j] += __shfl_xor(q[j], m, 32);
    }
  }
  if (lrow == 0) {   // lanes 0 and 16 publish their 8 rows each
#pragma unroll
    for (int j = 0; j < 8; ++j) {
      int row = mt * 16 + j + halfsel * 8;
      red[row][wn][0] = s[j];
      red[row][wn][1] = q[j];
    }
  }
  __syncthreads();

  if (tid < ROWS) {
    float su = red[tid][0][0] + red[tid][1][0] + red[tid][2][0] + red[tid][3][0];
    float sq = red[tid][0][1] + red[tid][1][1] + red[tid][2][1] + red[tid][3][1];
    float mean = su * (1.f / (float)NCOLS);
    float var  = sq * (1.f / (float)NCOLS) - mean * mean;
    stats[tid][0] = mean;
    stats[tid][1] = rsqrtf(var + 1e-5f);
  }
  __syncthreads();

  float mean_j[8], rstd_j[8];
#pragma unroll
  for (int j = 0; j < 8; ++j) {
    int row = mt * 16 + j + halfsel * 8;
    mean_j[j] = stats[row][0];
    rstd_j[j] = stats[row][1];
  }

#pragma unroll
  for (int ii = 0; ii < 12; ++ii) {
    int col  = (wn + 4 * ii) * 16 + lrow;
    float gm = gamma[col];
    float bt = beta[col];
#pragma unroll
    for (int j = 0; j < 8; ++j) {
      int rowg = blk * ROWS + mt * 16 + j + halfsel * 8;
      out[(size_t)rowg * NCOLS + col] = (c[ii][j] - mean_j[j]) * rstd_j[j] * gm + bt;
    }
  }
}

extern "C" void kernel_launch(void* const* d_in, const int* in_sizes, int n_in,
                              void* d_out, int out_size, void* d_ws, size_t ws_size,
                              hipStream_t stream) {
  (void)in_sizes; (void)n_in; (void)out_size; (void)ws_size;
  const float* X     = (const float*)d_in[0];
  const float* Wm    = (const float*)d_in[1];
  const float* bias  = (const float*)d_in[2];
  const float* gamma = (const float*)d_in[3];
  const float* beta  = (const float*)d_in[4];
  float* out         = (float*)d_out;

  uint16_t* Wth = (uint16_t*)d_ws;                       // 768*3840 bf16 = 5.9 MB
  uint16_t* Wtl = Wth + (size_t)NCOLS * KDIM;            // second plane

  // prep: 46080 threads = 180 blocks x 256
  hipLaunchKernelGGL(wprep_kernel, dim3(180), dim3(256), 0, stream, Wm, Wth, Wtl);

  // GEMM: 18432 rows / 64 per block = 288 blocks x 512 threads (16 wave32)
  hipLaunchKernelGGL(spt_fused_kernel, dim3(288), dim3(512), 0, stream,
                     X, Wth, Wtl, bias, gamma, beta, out);
}
